// DSSMEmbed_1417339208145
// MI455X (gfx1250) — matrix operand order repaired and checked by reading, not compiled, not run on hardware
//
#include <hip/hip_runtime.h>
#include <hip/hip_bf16.h>

typedef __attribute__((ext_vector_type(16))) __bf16 v16bf;
typedef __attribute__((ext_vector_type(8)))  float  v8f;
typedef __attribute__((ext_vector_type(4)))  unsigned int u32x4;
typedef __attribute__((ext_vector_type(8)))  int  i32x8;
typedef __attribute__((ext_vector_type(4)))  int  i32x4;

// ---------------------------------------------------------------------------
// Problem constants
// ---------------------------------------------------------------------------
#define BATCH 8192
#define HW    25          // 5x5
#define SE    32
#define CC    64
#define EDIM  256
#define NZ    512
#define FLAT  800         // 32*5*5

// ---------------------------------------------------------------------------
// Kernel 1: normalize z_vectors -> zn (f32) and zn_bf16
// ---------------------------------------------------------------------------
__global__ __launch_bounds__(256) void k_norm_z(const float* __restrict__ z,
                                                float* __restrict__ zn,
                                                __bf16* __restrict__ znb) {
  __shared__ float red[256];
  const int t = threadIdx.x;
  const int r = blockIdx.x;
  float v = z[(size_t)r * EDIM + t];
  red[t] = v * v;
  __syncthreads();
  for (int s = 128; s > 0; s >>= 1) {
    if (t < s) red[t] += red[t + s];
    __syncthreads();
  }
  float inv = 1.0f / sqrtf(red[0]);        // reference uses eps = 0 here
  float nv = v * inv;
  zn[(size_t)r * EDIM + t]  = nv;
  znb[(size_t)r * EDIM + t] = (__bf16)nv;
}

// ---------------------------------------------------------------------------
// Kernel 2: fused per-image pipeline (embedding/renorm, conv_embed on s and
// diff, both phi conv chains + linears + L2 norm, all in LDS).
// Writes embed1 as bf16 (GEMM A) and embed2 as f32 (argmax input).
// ---------------------------------------------------------------------------
__global__ __launch_bounds__(256) void k_phi(
    const int* __restrict__ s, const int* __restrict__ sp,
    const float* __restrict__ table,
    const float* __restrict__ cew, const float* __restrict__ ceb,
    const float* __restrict__ c1w1, const float* __restrict__ c1b1,
    const float* __restrict__ c2w1, const float* __restrict__ c2b1,
    const float* __restrict__ lw1,  const float* __restrict__ lb1,
    const float* __restrict__ c1w2, const float* __restrict__ c1b2,
    const float* __restrict__ c2w2, const float* __restrict__ c2b2,
    const float* __restrict__ lw2,  const float* __restrict__ lb2,
    float* __restrict__ embed2, __bf16* __restrict__ Abf) {
  __shared__ float rn[14];
  __shared__ int   sidx[HW], spidx[HW];
  __shared__ float embS[HW][SE], embD[HW][SE];
  __shared__ float xs[CC][HW], xd[CC][HW];
  __shared__ float h1[2][16][HW];
  __shared__ float h2[2][32][HW];
  __shared__ float red[256];

  const int t = threadIdx.x;
  const int b = blockIdx.x;

  if (t < 14) {
    float ss = 0.f;
    for (int c = 0; c < SE; ++c) { float v = table[t * SE + c]; ss += v * v; }
    rn[t] = fminf(1.f, 1.f / (sqrtf(ss) + 1e-7f));
  }
  if (t < HW) { sidx[t] = s[(size_t)b * HW + t]; spidx[t] = sp[(size_t)b * HW + t]; }
  __syncthreads();

  for (int e = t; e < HW * SE; e += 256) {
    int pix = e >> 5, c = e & 31;
    int rs = sidx[pix], rp = spidx[pix];
    float vs = table[rs * SE + c] * rn[rs];
    float vp = table[rp * SE + c] * rn[rp];
    embS[pix][c] = vs;
    embD[pix][c] = vp - vs;
  }
  __syncthreads();

  // ---- conv_embed (SE->CC), thread = (cc, q)
  {
    const int cc = t >> 2, q = t & 3;
    float accS[7], accD[7];
    int   pixs[7];
    int   np = 0;
    for (int p = q; p < HW; p += 4) { pixs[np] = p; accS[np] = ceb[cc]; accD[np] = 0.f; ++np; }
    for (int ky = 0; ky < 3; ++ky)
      for (int kx = 0; kx < 3; ++kx)
        for (int ic = 0; ic < SE; ++ic) {
          float w = cew[(cc * SE + ic) * 9 + ky * 3 + kx];
          for (int i = 0; i < np; ++i) {
            int p = pixs[i];
            int iy = p / 5 + ky - 1, ix = p % 5 + kx - 1;
            if (iy >= 0 && iy < 5 && ix >= 0 && ix < 5) {
              int ip = iy * 5 + ix;
              accS[i] += w * embS[ip][ic];
              accD[i] += w * embD[ip][ic];
            }
          }
        }
    for (int i = 0; i < np; ++i) { xs[cc][pixs[i]] = accS[i]; xd[cc][pixs[i]] = accD[i]; }
  }
  __syncthreads();

  // ---- conv1 (64->16), both phis: thread = (which, oc, q)
  {
    const int which = t >> 7, oc = (t >> 3) & 15, q = t & 7;
    const float* w_ = which ? c1w2 : c1w1;
    const float* b_ = which ? c1b2 : c1b1;
    float (*src)[HW] = which ? xd : xs;
    float acc[4]; int pixs[4]; int np = 0;
    for (int p = q; p < HW; p += 8) { pixs[np] = p; acc[np] = b_[oc]; ++np; }
    for (int ky = 0; ky < 3; ++ky)
      for (int kx = 0; kx < 3; ++kx)
        for (int ic = 0; ic < CC; ++ic) {
          float w = w_[(oc * CC + ic) * 9 + ky * 3 + kx];
          for (int i = 0; i < np; ++i) {
            int p = pixs[i];
            int iy = p / 5 + ky - 1, ix = p % 5 + kx - 1;
            if (iy >= 0 && iy < 5 && ix >= 0 && ix < 5)
              acc[i] += w * src[ic][iy * 5 + ix];
          }
        }
    for (int i = 0; i < np; ++i) h1[which][oc][pixs[i]] = fmaxf(acc[i], 0.f);
  }
  __syncthreads();

  // ---- conv2 (16->32), both phis: thread = (which, oc, q)
  {
    const int which = t >> 7, oc = (t >> 2) & 31, q = t & 3;
    const float* w_ = which ? c2w2 : c2w1;
    const float* b_ = which ? c2b2 : c2b1;
    float acc[7]; int pixs[7]; int np = 0;
    for (int p = q; p < HW; p += 4) { pixs[np] = p; acc[np] = b_[oc]; ++np; }
    for (int ky = 0; ky < 3; ++ky)
      for (int kx = 0; kx < 3; ++kx)
        for (int ic = 0; ic < 16; ++ic) {
          float w = w_[(oc * 16 + ic) * 9 + ky * 3 + kx];
          for (int i = 0; i < np; ++i) {
            int p = pixs[i];
            int iy = p / 5 + ky - 1, ix = p % 5 + kx - 1;
            if (iy >= 0 && iy < 5 && ix >= 0 && ix < 5)
              acc[i] += w * h1[which][ic][iy * 5 + ix];
          }
        }
    for (int i = 0; i < np; ++i) h2[which][oc][pixs[i]] = fmaxf(acc[i], 0.f);
  }
  __syncthreads();

  // ---- linear 800->256
  float a1 = lb1[t], a2 = lb2[t];
  const float* h2f0 = &h2[0][0][0];
  const float* h2f1 = &h2[1][0][0];
  for (int j = 0; j < FLAT; ++j) {
    a1 += h2f0[j] * lw1[(size_t)t * FLAT + j];
    a2 += h2f1[j] * lw2[(size_t)t * FLAT + j];
  }

  // ---- L2 normalize (eps = 1e-4)
  red[t] = a1 * a1; __syncthreads();
  for (int sft = 128; sft > 0; sft >>= 1) { if (t < sft) red[t] += red[t + sft]; __syncthreads(); }
  float n1 = sqrtf(red[0]); __syncthreads();
  red[t] = a2 * a2; __syncthreads();
  for (int sft = 128; sft > 0; sft >>= 1) { if (t < sft) red[t] += red[t + sft]; __syncthreads(); }
  float n2 = sqrtf(red[0]);

  Abf[(size_t)b * EDIM + t]    = (__bf16)(a1 / (n1 + 1e-4f));
  embed2[(size_t)b * EDIM + t] = a2 / (n2 + 1e-4f);
}

// ---------------------------------------------------------------------------
// Kernel 3: cosine argmax over 512 codebook rows + gather winner as bf16 B row
// ---------------------------------------------------------------------------
__global__ __launch_bounds__(256) void k_argmax(const float* __restrict__ embed2,
                                                const float* __restrict__ zn,
                                                const __bf16* __restrict__ znb,
                                                __bf16* __restrict__ Bzb) {
  __shared__ float ev[256];
  __shared__ float bval[256];
  __shared__ int   bidx[256];
  const int t = threadIdx.x, b = blockIdx.x;
  ev[t] = embed2[(size_t)b * EDIM + t];
  __syncthreads();
  float best = -1e30f; int bi = 0x7fffffff;
  for (int zz = t; zz < NZ; zz += 256) {
    const float* zr = zn + (size_t)zz * EDIM;
    float d = 0.f;
    for (int k = 0; k < EDIM; ++k) d += ev[k] * zr[k];
    if (d > best) { best = d; bi = zz; }   // first (lower) index wins ties
  }
  bval[t] = best; bidx[t] = bi;
  __syncthreads();
  for (int sft = 128; sft > 0; sft >>= 1) {
    if (t < sft) {
      float ov = bval[t + sft]; int oi = bidx[t + sft];
      if (ov > bval[t] || (ov == bval[t] && oi < bidx[t])) { bval[t] = ov; bidx[t] = oi; }
    }
    __syncthreads();
  }
  const int win = bidx[0];
  Bzb[(size_t)b * EDIM + t] = znb[(size_t)win * EDIM + t];
}

// ---------------------------------------------------------------------------
// Kernel 4: C[8192x8192] = exp(scale) * A(bf16) @ Bz(bf16)^T   (K = 256)
// 128x128 C-tile / block, 8 waves; wave tile 32x64 = 2x4 WMMA 16x16 tiles.
//
// Global->LDS staging via the Tensor Data Mover (double buffered):
//   D# describes a 32x128 bf16 tile of a [8192,256] row-major tensor
//   (data_size=2B, tile_dim0=32, tile_dim1=128, tensor_dim0_stride=256).
//   pad_enable inserts 4 DWORDs (16 B) after every 16 DWORDs (64 B = one
//   tile row), so LDS rows are 80 B (40 bf16) -> conflict-free b128 reads
//   (row stride 20 banks; gcd(20,64)=4 -> 16 distinct banks over 16 rows).
// Wave 0 issues the next step's two tensor loads, waits TENSORcnt<=2 so the
// current step's tiles are resident, then all 8 waves barrier and run the
// 8-WMMA burst while the TDM streams the next tiles.
// ---------------------------------------------------------------------------
#define BM 128
#define BN 128
#define BK 32
#define LSTR 40   // bf16 per LDS row (32 + 8 pad) == 80 B, matches TDM padding
#define NKSTEP (EDIM / BK)

__device__ __forceinline__ void tdm_load_tile(const __bf16* gsrc, unsigned lds_off) {
  const unsigned long long ga = (unsigned long long)(uintptr_t)gsrc;
  u32x4 g0;
  g0[0] = 1u;                                               // count=1, user mode
  g0[1] = lds_off;                                          // LDS byte address
  g0[2] = (unsigned)(ga & 0xffffffffull);                   // global_addr[31:0]
  g0[3] = (unsigned)((ga >> 32) & 0x1ffffffull) | (2u << 30); // ga[56:32] | type=2
  i32x8 g1;
  g1[0] = (1 << 16)        // data_size = 2 bytes
        | (1 << 20)        // pad_enable
        | (3 << 22)        // pad_interval: 16 DWORDs (= 64 B tile row)
        | (3 << 25);       // pad_amount:   4 DWORDs (= 16 B)
  g1[1] = (int)((unsigned)(EDIM & 0xffff) << 16);   // tensor_dim0 = 256 (lo16)
  g1[2] = (int)((unsigned)(BATCH & 0xffff) << 16);  // dim0 hi=0 | tensor_dim1 lo16
  g1[3] = (int)((unsigned)BK << 16);                // dim1 hi=0 | tile_dim0 = 32
  g1[4] = BM;                                       // tile_dim1 = 128 | tile_dim2=0
  g1[5] = EDIM;                                     // tensor_dim0_stride = 256
  g1[6] = 0;
  g1[7] = 0;
  const i32x4 z4 = {0, 0, 0, 0};
  const i32x8 z8 = {0, 0, 0, 0, 0, 0, 0, 0};
  // 6-arg form (clang-23 / therock-10.0 headers): groups 2,3 + extra group, cpol
  __builtin_amdgcn_tensor_load_to_lds(g0, g1, z4, z4, z8, 0);
}

__global__ __launch_bounds__(256) void k_gemm(const __bf16* __restrict__ A,
                                              const __bf16* __restrict__ Bz,
                                              const float* __restrict__ scale,
                                              float* __restrict__ C) {
  __shared__ __align__(16) __bf16 Al[2][BM * LSTR];
  __shared__ __align__(16) __bf16 Bl[2][BN * LSTR];

  const int t    = threadIdx.x;
  const int lane = t & 31;
  const int wave = t >> 5;
  const int wm   = wave & 3;   // 0..3 -> 32-row strip
  const int wn   = wave >> 2;  // 0..1 -> 64-col strip
  const int bm   = blockIdx.y * BM;
  const int bn   = blockIdx.x * BN;
  const int half = lane >> 4;
  const int l16  = lane & 15;

  const float es = expf(scale[0]);

  const unsigned ldsA[2] = {(unsigned)(size_t)&Al[0][0], (unsigned)(size_t)&Al[1][0]};
  const unsigned ldsB[2] = {(unsigned)(size_t)&Bl[0][0], (unsigned)(size_t)&Bl[1][0]};

  v8f acc[2][4];
  for (int mi = 0; mi < 2; ++mi)
    for (int ni = 0; ni < 4; ++ni)
      acc[mi][ni] = (v8f){0.f, 0.f, 0.f, 0.f, 0.f, 0.f, 0.f, 0.f};

  // prologue: stage K-step 0 into buffer 0
  if (wave == 0) {
    tdm_load_tile(A  + (size_t)bm * EDIM, ldsA[0]);
    tdm_load_tile(Bz + (size_t)bn * EDIM, ldsB[0]);
  }

  for (int ksi = 0; ksi < NKSTEP; ++ksi) {
    const int buf = ksi & 1;
    const int nxt = buf ^ 1;
    __syncthreads();                       // spare buffer free (prev compute done)
    if (wave == 0) {
      if (ksi + 1 < NKSTEP) {
        tdm_load_tile(A  + (size_t)bm * EDIM + (ksi + 1) * BK, ldsA[nxt]);
        tdm_load_tile(Bz + (size_t)bn * EDIM + (ksi + 1) * BK, ldsB[nxt]);
        __builtin_amdgcn_s_wait_tensorcnt(2);  // current step's 2 tiles resident
      } else {
        __builtin_amdgcn_s_wait_tensorcnt(0);
      }
    }
    __syncthreads();                       // release all waves onto `buf`

    v16bf af[2], bfr[4];
    for (int mi = 0; mi < 2; ++mi) {
      // A frag (16x32): lane<16 -> K {0..7,16..23}; lane>=16 -> K {8..15,24..31}
      const __bf16* p = &Al[buf][(wm * 32 + mi * 16 + l16) * LSTR + half * 8];
      union { v16bf v; uint4 u[2]; } ua;
      ua.u[0] = *(const uint4*)p;
      ua.u[1] = *(const uint4*)(p + 16);
      af[mi] = ua.v;
    }
    for (int ni = 0; ni < 4; ++ni) {
      // B frag (32x16): lane<16 -> K {0..7,16..23}; lane>=16 -> K {8..15,24..31}
      const __bf16* p = &Bl[buf][(wn * 64 + ni * 16 + l16) * LSTR + half * 8];
      union { v16bf v; uint4 u[2]; } ub;
      ub.u[0] = *(const uint4*)p;
      ub.u[1] = *(const uint4*)(p + 16);
      bfr[ni] = ub.v;
    }
    for (int mi = 0; mi < 2; ++mi)
      for (int ni = 0; ni < 4; ++ni)
        acc[mi][ni] = __builtin_amdgcn_wmma_f32_16x16x32_bf16(
            false, af[mi], false, bfr[ni], (short)0, acc[mi][ni], false, false);
  }

  // store: C/D layout -> vgpr r: lanes 0-15 row M=r, lanes 16-31 row M=8+r
  for (int mi = 0; mi < 2; ++mi)
    for (int ni = 0; ni < 4; ++ni) {
      const int row0 = bm + wm * 32 + mi * 16 + half * 8;
      const int col  = bn + wn * 64 + ni * 16 + l16;
      for (int r = 0; r < 8; ++r)
        C[(size_t)(row0 + r) * BATCH + col] = es * acc[mi][ni][r];
    }
}

// ---------------------------------------------------------------------------
// Launch
// ---------------------------------------------------------------------------
extern "C" void kernel_launch(void* const* d_in, const int* in_sizes, int n_in,
                              void* d_out, int out_size, void* d_ws, size_t ws_size,
                              hipStream_t stream) {
  (void)in_sizes; (void)n_in; (void)out_size; (void)ws_size;

  const int*   s      = (const int*)d_in[0];
  const int*   sp     = (const int*)d_in[1];
  const float* table  = (const float*)d_in[2];
  const float* cew    = (const float*)d_in[3];
  const float* ceb    = (const float*)d_in[4];
  const float* c1w1   = (const float*)d_in[5];
  const float* c1b1   = (const float*)d_in[6];
  const float* c2w1   = (const float*)d_in[7];
  const float* c2b1   = (const float*)d_in[8];
  const float* lw1    = (const float*)d_in[9];
  const float* lb1    = (const float*)d_in[10];
  const float* c1w2   = (const float*)d_in[11];
  const float* c1b2   = (const float*)d_in[12];
  const float* c2w2   = (const float*)d_in[13];
  const float* c2b2   = (const float*)d_in[14];
  const float* lw2    = (const float*)d_in[15];
  const float* lb2    = (const float*)d_in[16];
  const float* zvec   = (const float*)d_in[17];
  const float* scale  = (const float*)d_in[18];

  char* ws = (char*)d_ws;
  size_t off = 0;
  float*  zn     = (float*)(ws + off);  off += (size_t)NZ * EDIM * sizeof(float);      // 512 KB
  __bf16* znb    = (__bf16*)(ws + off); off += (size_t)NZ * EDIM * sizeof(__bf16);     // 256 KB
  float*  embed2 = (float*)(ws + off);  off += (size_t)BATCH * EDIM * sizeof(float);   // 8 MB
  __bf16* Abf    = (__bf16*)(ws + off); off += (size_t)BATCH * EDIM * sizeof(__bf16);  // 4 MB
  __bf16* Bzb    = (__bf16*)(ws + off); off += (size_t)BATCH * EDIM * sizeof(__bf16);  // 4 MB

  k_norm_z<<<NZ, 256, 0, stream>>>(zvec, zn, znb);

  k_phi<<<BATCH, 256, 0, stream>>>(s, sp, table, cew, ceb,
                                   c1w1, c1b1, c2w1, c2b1, lw1, lb1,
                                   c1w2, c1b2, c2w2, c2b2, lw2, lb2,
                                   embed2, Abf);

  k_argmax<<<BATCH, 256, 0, stream>>>(embed2, zn, znb, Bzb);

  dim3 grid(BATCH / BN, BATCH / BM);
  k_gemm<<<grid, 256, 0, stream>>>(Abf, Bzb, scale, (float*)d_out);
}
